// CrystalGraphEncoder_44427141710552
// MI455X (gfx1250) — matrix operand order repaired and checked by reading, not compiled, not run on hardware
//
#include <hip/hip_runtime.h>
#include <hip/hip_bf16.h>

// ---------------------------------------------------------------------------
// CrystalGraphEncoder for MI455X (gfx1250): bf16 WMMA GEMMs + LDS staging.
// ---------------------------------------------------------------------------

#define NATOMS 10000
#define NEDGES 160000
#define HID    256
#define EMB    128
#define NLAY   4
#define NGRAPH 256

typedef __attribute__((ext_vector_type(16))) __bf16 v16bf;
typedef __attribute__((ext_vector_type(8)))  float  v8f;

struct __align__(16) U4 { unsigned int a, b, c, d; };
union BF16x16 { U4 q[2]; v16bf v; };

static __device__ inline unsigned short f32_to_bf16(float f) {
    unsigned int u = __float_as_uint(f);
    u = (u + 0x7FFFu + ((u >> 16) & 1u)) >> 16;           // round-to-nearest-even
    return (unsigned short)u;
}
static __device__ inline float bf16_to_f32(unsigned short s) {
    return __uint_as_float(((unsigned int)s) << 16);
}
static __device__ inline void atomicAddF(float* p, float v) {
    __hip_atomic_fetch_add(p, v, __ATOMIC_RELAXED, __HIP_MEMORY_SCOPE_AGENT);
}
static __device__ inline v8f wmma_bf16(v16bf a, v16bf b, v8f c) {
    return __builtin_amdgcn_wmma_f32_16x16x32_bf16(false, a, false, b, (short)0, c, false, false);
}

// ---------------------------------------------------------------------------
// Repack a row-major f32 weight matrix W[K][N] into WMMA-B bf16 blocks:
// element index = (((ntile*KT + ktile)*32 + lane)*16 + e), with
//   n = ntile*16 + lane%16 ;  k = ktile*32 + (lane/16)*8 + (e%8) + (e/8)*16
// so each lane's 16 bf16 B-fragment elements are contiguous (2x b128 loads).
// ---------------------------------------------------------------------------
__global__ void repack_b_kernel(const float* __restrict__ src,
                                unsigned short* __restrict__ dst, int K, int N) {
    int t = blockIdx.x * blockDim.x + threadIdx.x;
    if (t >= K * N) return;
    int kt_count = K >> 5;
    int blk  = t >> 9;
    int r    = t & 511;
    int lane = r >> 4;
    int e    = r & 15;
    int ktile = blk % kt_count;
    int ntile = blk / kt_count;
    int n = ntile * 16 + (lane & 15);
    int k = ktile * 32 + (lane >> 4) * 8 + (e & 7) + (e >> 3) * 16;
    dst[t] = f32_to_bf16(src[k * N + n]);
}

// ---------------------------------------------------------------------------
// x[i][c] = elem_emb[atomic_numbers[i]-1][c]
// ---------------------------------------------------------------------------
__global__ void embed_kernel(const int* __restrict__ an,
                             const float* __restrict__ emb,
                             float* __restrict__ x, int n) {
    int i = blockIdx.x * blockDim.x + threadIdx.x;
    if (i >= n * HID) return;
    int row = i >> 8, c = i & 255;
    x[i] = emb[(an[row] - 1) * HID + c];
}

// ---------------------------------------------------------------------------
// Edge MLP + message scatter. 32 edges / block, 16 waves (512 thr).
//   edge_in = [x[src] | x[dst] | edge_attr]  (edge_attr recomputed: 10-dim dot)
//   h = relu(edge_in @ ew1 + eb1)       K=768  (24 WMMA k-steps x 2 M-tiles)
//   gate = sigmoid(h @ ew2 + eb2)       K=256  ( 8 WMMA k-steps x 2 M-tiles)
//   agg[dst] += x[src] * gate           (f32 global atomics)
// LDS: edge_in 32x768 bf16 = 48KB; h overlays the dead edge_attr columns.
// ---------------------------------------------------------------------------
#define EKT1 24
#define EKT2 8
__global__ __launch_bounds__(512) void edge_mlp_kernel(
    const float* __restrict__ x,
    const int* __restrict__ edge_index,
    const float* __restrict__ edge_features,
    const float* __restrict__ edge_w,
    const float* __restrict__ edge_b,
    const unsigned short* __restrict__ ew1r, const float* __restrict__ eb1,
    const unsigned short* __restrict__ ew2r, const float* __restrict__ eb2,
    float* __restrict__ agg, int n_edges)
{
    __shared__ __align__(16) unsigned short lin[32][768];
    __shared__ int   lsrc[32];
    __shared__ int   ldst[32];
    __shared__ float lef[32][10];

    const int tid = threadIdx.x;
    const int e0  = blockIdx.x * 32;

    if (tid < 32) {
        int e = e0 + tid; if (e >= n_edges) e = n_edges - 1;
        lsrc[tid] = edge_index[e];
        ldst[tid] = edge_index[n_edges + e];
    }
    for (int i = tid; i < 320; i += 512) {
        int row = i / 10, b = i % 10;
        int e = e0 + row; if (e >= n_edges) e = n_edges - 1;
        lef[row][b] = edge_features[e * 10 + b];
    }
    __syncthreads();

    for (int i = tid; i < 32 * 768; i += 512) {
        int row = i / 768, c = i % 768;
        float v;
        if (c < 256)       v = x[lsrc[row] * HID + c];
        else if (c < 512)  v = x[ldst[row] * HID + (c - 256)];
        else {
            int cc = c - 512;
            v = edge_b[cc];
            #pragma unroll
            for (int b = 0; b < 10; ++b) v += lef[row][b] * edge_w[b * HID + cc];
        }
        lin[row][c] = f32_to_bf16(v);
    }
    __syncthreads();

    const int wave = tid >> 5, lane = tid & 31;
    const int lm = lane & 15, lh = lane >> 4;

    v8f acc0 = {0.f,0.f,0.f,0.f,0.f,0.f,0.f,0.f};
    v8f acc1 = acc0;
    BF16x16 A, B;

    const U4* bw1 = (const U4*)ew1r;
    for (int kt = 0; kt < EKT1; ++kt) {
        const U4* bp = bw1 + ((((wave * EKT1) + kt) * 32 + lane) << 1);
        B.q[0] = bp[0]; B.q[1] = bp[1];
        int k0 = kt * 32 + lh * 8;
        A.q[0] = *(const U4*)&lin[lm][k0];
        A.q[1] = *(const U4*)&lin[lm][k0 + 16];
        acc0 = wmma_bf16(A.v, B.v, acc0);
        A.q[0] = *(const U4*)&lin[16 + lm][k0];
        A.q[1] = *(const U4*)&lin[16 + lm][k0 + 16];
        acc1 = wmma_bf16(A.v, B.v, acc1);
    }

    const int col = wave * 16 + lm;
    const float b1v = eb1[col];
    #pragma unroll
    for (int r = 0; r < 8; ++r) {
        lin[lh * 8 + r][512 + col]      = f32_to_bf16(fmaxf(acc0[r] + b1v, 0.f));
        lin[16 + lh * 8 + r][512 + col] = f32_to_bf16(fmaxf(acc1[r] + b1v, 0.f));
    }
    __syncthreads();

    v8f g0 = {0.f,0.f,0.f,0.f,0.f,0.f,0.f,0.f};
    v8f g1 = g0;
    const U4* bw2 = (const U4*)ew2r;
    for (int kt = 0; kt < EKT2; ++kt) {
        const U4* bp = bw2 + ((((wave * EKT2) + kt) * 32 + lane) << 1);
        B.q[0] = bp[0]; B.q[1] = bp[1];
        int k0 = 512 + kt * 32 + lh * 8;
        A.q[0] = *(const U4*)&lin[lm][k0];
        A.q[1] = *(const U4*)&lin[lm][k0 + 16];
        g0 = wmma_bf16(A.v, B.v, g0);
        A.q[0] = *(const U4*)&lin[16 + lm][k0];
        A.q[1] = *(const U4*)&lin[16 + lm][k0 + 16];
        g1 = wmma_bf16(A.v, B.v, g1);
    }

    const float b2v = eb2[col];
    #pragma unroll
    for (int r = 0; r < 8; ++r) {
        int row0 = lh * 8 + r;
        float gate0 = 1.f / (1.f + __expf(-(g0[r] + b2v)));
        if (e0 + row0 < n_edges) {
            float sf = bf16_to_f32(lin[row0][col]);
            atomicAddF(&agg[ldst[row0] * HID + col], sf * gate0);
        }
        int row1 = 16 + row0;
        float gate1 = 1.f / (1.f + __expf(-(g1[r] + b2v)));
        if (e0 + row1 < n_edges) {
            float sf = bf16_to_f32(lin[row1][col]);
            atomicAddF(&agg[ldst[row1] * HID + col], sf * gate1);
        }
    }
}

// ---------------------------------------------------------------------------
// Node MLP: u = relu([x|agg] @ nw1 + nb1) @ nw2 + nb2.  32 nodes / block.
// ---------------------------------------------------------------------------
#define NKT1 16
#define NKT2 8
__global__ __launch_bounds__(512) void node_mlp_kernel(
    const float* __restrict__ x, const float* __restrict__ agg,
    const unsigned short* __restrict__ nw1r, const float* __restrict__ nb1,
    const unsigned short* __restrict__ nw2r, const float* __restrict__ nb2,
    float* __restrict__ u, int n_nodes)
{
    __shared__ __align__(16) unsigned short lin[32][512];
    __shared__ __align__(16) unsigned short lh_[32][256];

    const int tid = threadIdx.x;
    const int r0  = blockIdx.x * 32;

    for (int i = tid; i < 32 * 512; i += 512) {
        int row = i >> 9, c = i & 511;
        int gr = r0 + row; if (gr >= n_nodes) gr = n_nodes - 1;
        float v = (c < 256) ? x[gr * HID + c] : agg[gr * HID + (c - 256)];
        lin[row][c] = f32_to_bf16(v);
    }
    __syncthreads();

    const int wave = tid >> 5, lane = tid & 31;
    const int lm = lane & 15, lh = lane >> 4;

    v8f acc0 = {0.f,0.f,0.f,0.f,0.f,0.f,0.f,0.f};
    v8f acc1 = acc0;
    BF16x16 A, B;

    const U4* bw1 = (const U4*)nw1r;
    for (int kt = 0; kt < NKT1; ++kt) {
        const U4* bp = bw1 + ((((wave * NKT1) + kt) * 32 + lane) << 1);
        B.q[0] = bp[0]; B.q[1] = bp[1];
        int k0 = kt * 32 + lh * 8;
        A.q[0] = *(const U4*)&lin[lm][k0];
        A.q[1] = *(const U4*)&lin[lm][k0 + 16];
        acc0 = wmma_bf16(A.v, B.v, acc0);
        A.q[0] = *(const U4*)&lin[16 + lm][k0];
        A.q[1] = *(const U4*)&lin[16 + lm][k0 + 16];
        acc1 = wmma_bf16(A.v, B.v, acc1);
    }

    const int col = wave * 16 + lm;
    const float b1v = nb1[col];
    #pragma unroll
    for (int r = 0; r < 8; ++r) {
        lh_[lh * 8 + r][col]      = f32_to_bf16(fmaxf(acc0[r] + b1v, 0.f));
        lh_[16 + lh * 8 + r][col] = f32_to_bf16(fmaxf(acc1[r] + b1v, 0.f));
    }
    __syncthreads();

    v8f u0 = {0.f,0.f,0.f,0.f,0.f,0.f,0.f,0.f};
    v8f u1 = u0;
    const U4* bw2 = (const U4*)nw2r;
    for (int kt = 0; kt < NKT2; ++kt) {
        const U4* bp = bw2 + ((((wave * NKT2) + kt) * 32 + lane) << 1);
        B.q[0] = bp[0]; B.q[1] = bp[1];
        int k0 = kt * 32 + lh * 8;
        A.q[0] = *(const U4*)&lh_[lm][k0];
        A.q[1] = *(const U4*)&lh_[lm][k0 + 16];
        u0 = wmma_bf16(A.v, B.v, u0);
        A.q[0] = *(const U4*)&lh_[16 + lm][k0];
        A.q[1] = *(const U4*)&lh_[16 + lm][k0 + 16];
        u1 = wmma_bf16(A.v, B.v, u1);
    }

    const float b2v = nb2[col];
    #pragma unroll
    for (int r = 0; r < 8; ++r) {
        int row0 = lh * 8 + r, row1 = 16 + row0;
        if (r0 + row0 < n_nodes) u[(r0 + row0) * HID + col] = u0[r] + b2v;
        if (r0 + row1 < n_nodes) u[(r0 + row1) * HID + col] = u1[r] + b2v;
    }
}

// ---------------------------------------------------------------------------
// BatchNorm (training stats, biased var) + residual add.
// ---------------------------------------------------------------------------
__global__ void bn_stats_kernel(const float* __restrict__ u,
                                float* __restrict__ mu, float* __restrict__ rs, int n) {
    int c = blockIdx.x, tid = threadIdx.x;
    float s = 0.f, s2 = 0.f;
    for (int i = tid; i < n; i += 256) {
        float v = u[i * HID + c];
        s += v; s2 += v * v;
    }
    __shared__ float sh[256], sh2[256];
    sh[tid] = s; sh2[tid] = s2;
    __syncthreads();
    for (int o = 128; o > 0; o >>= 1) {
        if (tid < o) { sh[tid] += sh[tid + o]; sh2[tid] += sh2[tid + o]; }
        __syncthreads();
    }
    if (tid == 0) {
        float m   = sh[0] / (float)n;
        float var = sh2[0] / (float)n - m * m;
        mu[c] = m;
        rs[c] = rsqrtf(var + 1e-5f);
    }
}

__global__ void bn_apply_kernel(float* __restrict__ x, const float* __restrict__ u,
                                const float* __restrict__ mu, const float* __restrict__ rs,
                                const float* __restrict__ gamma, const float* __restrict__ beta,
                                int n) {
    int i = blockIdx.x * blockDim.x + threadIdx.x;
    if (i >= n * HID) return;
    int c = i & 255;
    x[i] += (u[i] - mu[c]) * rs[c] * gamma[c] + beta[c];
}

// ---------------------------------------------------------------------------
// Graph mean pooling + readout MLP + L2 normalize.
// ---------------------------------------------------------------------------
__global__ void pool_sum_kernel(const float* __restrict__ x, const int* __restrict__ batch,
                                float* __restrict__ pooled, int n) {
    int i = blockIdx.x * blockDim.x + threadIdx.x;
    if (i >= n * HID) return;
    int row = i >> 8;
    atomicAddF(&pooled[batch[row] * HID + (i & 255)], x[i]);
}
__global__ void pool_count_kernel(const int* __restrict__ batch, float* __restrict__ counts, int n) {
    int i = blockIdx.x * blockDim.x + threadIdx.x;
    if (i >= n) return;
    atomicAddF(&counts[batch[i]], 1.f);
}
__global__ void pool_div_kernel(float* __restrict__ pooled, const float* __restrict__ counts) {
    int i = blockIdx.x * blockDim.x + threadIdx.x;
    if (i >= NGRAPH * HID) return;
    pooled[i] /= fmaxf(counts[i >> 8], 1.f);
}
__global__ void ro1_kernel(const float* __restrict__ pooled, const float* __restrict__ w1,
                           const float* __restrict__ b1, float* __restrict__ t) {
    int i = blockIdx.x * blockDim.x + threadIdx.x;   // NGRAPH*HID
    int g = i >> 8, j = i & 255;
    float s = b1[j];
    for (int k = 0; k < HID; ++k) s += pooled[g * HID + k] * w1[k * HID + j];
    t[i] = fmaxf(s, 0.f);
}
__global__ void ro2_kernel(const float* __restrict__ t, const float* __restrict__ w2,
                           const float* __restrict__ b2, float* __restrict__ emb) {
    int i = blockIdx.x * blockDim.x + threadIdx.x;   // NGRAPH*EMB
    int g = i >> 7, j = i & 127;
    float s = b2[j];
    for (int k = 0; k < HID; ++k) s += t[g * HID + k] * w2[k * EMB + j];
    emb[i] = s;
}
__global__ void norm_kernel(float* __restrict__ emb) {
    int g = blockIdx.x, j = threadIdx.x;             // 128 threads / graph
    float v = emb[g * EMB + j];
    __shared__ float sh[128];
    sh[j] = v * v;
    __syncthreads();
    for (int o = 64; o > 0; o >>= 1) {
        if (j < o) sh[j] += sh[j + o];
        __syncthreads();
    }
    emb[g * EMB + j] = v / fmaxf(sqrtf(sh[0]), 1e-12f);
}

// ---------------------------------------------------------------------------
extern "C" void kernel_launch(void* const* d_in, const int* in_sizes, int n_in,
                              void* d_out, int out_size, void* d_ws, size_t ws_size,
                              hipStream_t stream) {
    const int*   atomic_numbers = (const int*)  d_in[0];
    const int*   edge_index     = (const int*)  d_in[1];
    const float* edge_features  = (const float*)d_in[2];
    const int*   batch          = (const int*)  d_in[3];
    const float* elem_emb       = (const float*)d_in[4];
    const float* edge_w         = (const float*)d_in[5];
    const float* edge_b         = (const float*)d_in[6];
    const float* conv_ew1       = (const float*)d_in[7];
    const float* conv_eb1       = (const float*)d_in[8];
    const float* conv_ew2       = (const float*)d_in[9];
    const float* conv_eb2       = (const float*)d_in[10];
    const float* conv_nw1       = (const float*)d_in[11];
    const float* conv_nb1       = (const float*)d_in[12];
    const float* conv_nw2       = (const float*)d_in[13];
    const float* conv_nb2       = (const float*)d_in[14];
    const float* conv_gamma     = (const float*)d_in[15];
    const float* conv_beta      = (const float*)d_in[16];
    const float* ro_w1          = (const float*)d_in[17];
    const float* ro_b1          = (const float*)d_in[18];
    const float* ro_w2          = (const float*)d_in[19];
    const float* ro_b2          = (const float*)d_in[20];
    float* out = (float*)d_out;

    // workspace bump allocator (256B aligned)
    char* wp = (char*)d_ws;
    auto balloc = [&](size_t bytes) {
        char* p = wp;
        wp += (bytes + 255) & ~(size_t)255;
        return p;
    };
    float* x      = (float*)balloc((size_t)NATOMS * HID * 4);
    float* agg    = (float*)balloc((size_t)NATOMS * HID * 4);
    float* u      = (float*)balloc((size_t)NATOMS * HID * 4);
    float* mu     = (float*)balloc(HID * 4);
    float* rs     = (float*)balloc(HID * 4);
    float* pooled = (float*)balloc((size_t)NGRAPH * HID * 4);
    float* counts = (float*)balloc(NGRAPH * 4);
    float* tbuf   = (float*)balloc((size_t)NGRAPH * HID * 4);
    unsigned short* ew1r = (unsigned short*)balloc((size_t)NLAY * 768 * 256 * 2);
    unsigned short* ew2r = (unsigned short*)balloc((size_t)NLAY * 256 * 256 * 2);
    unsigned short* nw1r = (unsigned short*)balloc((size_t)NLAY * 512 * 256 * 2);
    unsigned short* nw2r = (unsigned short*)balloc((size_t)NLAY * 256 * 256 * 2);

    // ---- repack weights to WMMA-B bf16 layout
    for (int l = 0; l < NLAY; ++l) {
        repack_b_kernel<<<(768 * 256 + 255) / 256, 256, 0, stream>>>(
            conv_ew1 + (size_t)l * 768 * 256, ew1r + (size_t)l * 768 * 256, 768, 256);
        repack_b_kernel<<<(256 * 256 + 255) / 256, 256, 0, stream>>>(
            conv_ew2 + (size_t)l * 256 * 256, ew2r + (size_t)l * 256 * 256, 256, 256);
        repack_b_kernel<<<(512 * 256 + 255) / 256, 256, 0, stream>>>(
            conv_nw1 + (size_t)l * 512 * 256, nw1r + (size_t)l * 512 * 256, 512, 256);
        repack_b_kernel<<<(256 * 256 + 255) / 256, 256, 0, stream>>>(
            conv_nw2 + (size_t)l * 256 * 256, nw2r + (size_t)l * 256 * 256, 256, 256);
    }

    // ---- embedding lookup
    embed_kernel<<<(NATOMS * HID + 255) / 256, 256, 0, stream>>>(atomic_numbers, elem_emb, x, NATOMS);

    // ---- conv layers
    for (int l = 0; l < NLAY; ++l) {
        hipMemsetAsync(agg, 0, (size_t)NATOMS * HID * 4, stream);
        edge_mlp_kernel<<<NEDGES / 32, 512, 0, stream>>>(
            x, edge_index, edge_features, edge_w, edge_b,
            ew1r + (size_t)l * 768 * 256, conv_eb1 + l * HID,
            ew2r + (size_t)l * 256 * 256, conv_eb2 + l * HID,
            agg, NEDGES);
        node_mlp_kernel<<<(NATOMS + 31) / 32, 512, 0, stream>>>(
            x, agg,
            nw1r + (size_t)l * 512 * 256, conv_nb1 + l * HID,
            nw2r + (size_t)l * 256 * 256, conv_nb2 + l * HID,
            u, NATOMS);
        bn_stats_kernel<<<HID, 256, 0, stream>>>(u, mu, rs, NATOMS);
        bn_apply_kernel<<<(NATOMS * HID + 255) / 256, 256, 0, stream>>>(
            x, u, mu, rs, conv_gamma + l * HID, conv_beta + l * HID, NATOMS);
    }

    // ---- pooling + readout
    hipMemsetAsync(pooled, 0, (size_t)NGRAPH * HID * 4, stream);
    hipMemsetAsync(counts, 0, NGRAPH * 4, stream);
    pool_sum_kernel<<<(NATOMS * HID + 255) / 256, 256, 0, stream>>>(x, batch, pooled, NATOMS);
    pool_count_kernel<<<(NATOMS + 255) / 256, 256, 0, stream>>>(batch, counts, NATOMS);
    pool_div_kernel<<<(NGRAPH * HID + 255) / 256, 256, 0, stream>>>(pooled, counts);
    ro1_kernel<<<NGRAPH * HID / 256, 256, 0, stream>>>(pooled, ro_w1, ro_b1, tbuf);
    ro2_kernel<<<NGRAPH * EMB / 256, 256, 0, stream>>>(tbuf, ro_w2, ro_b2, out);
    norm_kernel<<<NGRAPH, EMB, 0, stream>>>(out);
}
